// VanillaRNN_29515015258162
// MI455X (gfx1250) — matrix operand order repaired and checked by reading, not compiled, not run on hardware
//
#include <hip/hip_runtime.h>
#include <hip/hip_bf16.h>

// Problem constants (match reference)
#define T_STEPS 512
#define BB      128
#define DD      256
#define HH      512
#define CC      10
#define KTOT    (DD + HH)        // 768
#define KPAD    (KTOT + 8)       // 776: row stride = 388 dwords == 4 mod 64 banks
#define NCHUNK  (KTOT / 32)      // 24 k-chunks of 32
#define NTILES  (HH / 16)        // 32 column tiles (one per wave)

typedef __attribute__((ext_vector_type(16))) __bf16       v16bf;
typedef __attribute__((ext_vector_type(8)))  float        v8f;
typedef __attribute__((ext_vector_type(4)))  float        v4f;
typedef __attribute__((ext_vector_type(4)))  unsigned int v4u;
typedef __attribute__((ext_vector_type(8)))  unsigned int v8u;
typedef __attribute__((ext_vector_type(2)))  unsigned int v2u;

static __device__ __forceinline__ unsigned int f32_to_bf16_bits(float f) {
    unsigned int u = __builtin_bit_cast(unsigned int, f);
    u += 0x7FFFu + ((u >> 16) & 1u);
    return u >> 16;
}
static __device__ __forceinline__ unsigned int pack_bf16x2(float a, float b) {
    return f32_to_bf16_bits(a) | (f32_to_bf16_bits(b) << 16);
}

// ---------------------------------------------------------------------------
// One-shot: pack [W_hx ; W_hh] (fp32, 768 x 512) into bf16 fragment-major
// layout matching the CDNA5 B-matrix (32x16) lane layout, 32 B/lane contiguous.
// ---------------------------------------------------------------------------
__global__ __launch_bounds__(256) void rnn_pack_weights(
    const float* __restrict__ W_hx, const float* __restrict__ W_hh,
    unsigned short* __restrict__ Wfrag)
{
    int idx     = blockIdx.x * 256 + threadIdx.x;   // 0 .. 24*32*512-1
    int j       = idx & 15;
    int lane    = (idx >> 4) & 31;
    int fragIdx = idx >> 9;                          // kc*NTILES + ntile
    int kc      = fragIdx >> 5;
    int ntile   = fragIdx & 31;
    int n       = (ntile << 4) + (lane & 15);
    int kb      = (lane >= 16) ? 8 : 0;
    int k       = kc * 32 + kb + (j & 7) + ((j & 8) << 1);
    float v = (k < DD) ? W_hx[k * HH + n] : W_hh[(k - DD) * HH + n];
    Wfrag[idx] = (unsigned short)f32_to_bf16_bits(v);
}

// ---------------------------------------------------------------------------
// Persistent recurrence kernel: 8 independent row-block recurrences.
//   One WG (1024 threads = 32 waves, one full WGP) owns 16 batch rows and ALL
//   512 hidden columns for all 512 timesteps. h state lives in the LDS A-panel
//   as bf16; tanh outputs are stored straight back as the next step's A h-part.
//   No cross-WG sync exists; per step = 2 hardware workgroup barriers.
//   NOTE: Wfrag/hfinal intentionally NOT restrict, plus a per-step compiler
//   memory barrier, so B fragment loads are NOT LICM'd into spilled registers.
// ---------------------------------------------------------------------------
__global__ __launch_bounds__(1024) void rnn_persist(
    const float* __restrict__ x,                   // [512,128,256]
    const unsigned short* Wfrag,                   // packed bf16 B fragments
    const float* __restrict__ b_h,                 // [512]
    float* hfinal)                                 // [128,512] fp32 (written at t=511)
{
    __shared__ __align__(16) unsigned short ldsA[16 * KPAD];   // ~24.3 KB

    const int tid   = threadIdx.x;
    const int lane  = tid & 31;
    const int wave  = tid >> 5;                     // 0..31 == ntile
    const int row0  = blockIdx.x * 16;              // row block (8 WGs)
    const int n0    = wave << 4;
    const int ln    = lane & 15;
    const int hi    = lane >> 4;

    const float bias = b_h[n0 + ln];
    // A-fragment LDS element offset (16-bit A layout; row = ln)
    //   lane<16: K 0..7 and 16..23 ; lane>=16: K 8..15 and 24..31
    const int arow = ln * KPAD + (hi << 3);
    const unsigned short* bbase = Wfrag + (size_t)wave * 512 + (size_t)lane * 16;

    // x staging geometry: 16 rows x 256 cols, one float4 per thread
    const int sm = tid >> 6;                        // 0..15
    const int sc = (tid & 63) << 2;                 // 0..252
    const float* xbase = x + ((size_t)row0 + sm) * DD + sc;   // advance by BB*DD per t

    // ---- h0 = 0: zero the LDS h-region (cols 256..767), 8 elems/thread ----
    {
        int e = tid << 3;                           // 0..8184
        int m = e >> 9, c = e & 511;
        v4u z = {0u, 0u, 0u, 0u};
        *(v4u*)&ldsA[m * KPAD + DD + c] = z;        // 16B-aligned ds_store_b128
    }

    // preload x(t=0)
    v4f xf = *(const v4f*)xbase;

#pragma unroll 1
    for (int t = 0; t < T_STEPS; ++t) {
        // defeat LICM/CSE of the loop-"invariant" B fragment loads
        asm volatile("" ::: "memory");

        // ---- stage preloaded x panel (registers -> bf16 LDS cols 0..255) ----
        {
            v2u p; p[0] = pack_bf16x2(xf[0], xf[1]); p[1] = pack_bf16x2(xf[2], xf[3]);
            *(v2u*)&ldsA[sm * KPAD + sc] = p;       // ds_store_b64, conflict-free
        }
        __syncthreads();   // A panel complete (x fresh, h from previous step)

        // ---- issue next step's x load + distance-2 L2 prefetch (overlaps WMMAs) ----
        {
            int tn = (t + 1 < T_STEPS) ? (t + 1) : t;
            const float* xp = xbase + (size_t)tn * BB * DD;
            xf = *(const v4f*)xp;                              // global_load_b128
            __builtin_prefetch(xp + (size_t)BB * DD, 0, 1);    // warm L2 for t+2
        }

        // ---- 24-deep WMMA stream: A from LDS, B streamed from L2 ----
        v8f c0a, c1a;
#pragma unroll
        for (int r = 0; r < 8; ++r) { c0a[r] = bias; c1a[r] = 0.0f; }

#pragma unroll
        for (int kc = 0; kc < NCHUNK; ++kc) {
            v4u alo = *(const v4u*)&ldsA[arow + kc * 32];        // ds_load_b128
            v4u ahi = *(const v4u*)&ldsA[arow + kc * 32 + 16];   // (2-pass, no conflicts)
            const v4u* bp = (const v4u*)(bbase + (size_t)kc * NTILES * 512);
            v4u blo = bp[0];                                     // global_load_b128 (L2-hot)
            v4u bhi = bp[1];

            v8u au, bu;
#pragma unroll
            for (int q = 0; q < 4; ++q) {
                au[q] = alo[q]; au[q + 4] = ahi[q];
                bu[q] = blo[q]; bu[q + 4] = bhi[q];
            }

            if (kc & 1)
                c1a = __builtin_amdgcn_wmma_f32_16x16x32_bf16(
                        false, __builtin_bit_cast(v16bf, au),
                        false, __builtin_bit_cast(v16bf, bu),
                        (short)0, c1a, false, false);
            else
                c0a = __builtin_amdgcn_wmma_f32_16x16x32_bf16(
                        false, __builtin_bit_cast(v16bf, au),
                        false, __builtin_bit_cast(v16bf, bu),
                        (short)0, c0a, false, false);
        }
        __syncthreads();   // all A reads done before anyone overwrites the panel

        // ---- tanh; write h straight back into the LDS A-panel (bf16) ----
#pragma unroll
        for (int r = 0; r < 8; ++r) {
            int m = r + (hi << 3);                  // local row 0..15
            float v = tanhf(c0a[r] + c1a[r]);
            ldsA[m * KPAD + DD + n0 + ln] = (unsigned short)f32_to_bf16_bits(v);
            if (t == T_STEPS - 1)
                hfinal[(row0 + m) * HH + n0 + ln] = v;   // fp32 h only at the end
        }
        // next iteration's __syncthreads (after x staging) publishes these stores
    }
}

// ---------------------------------------------------------------------------
// Output projection: out[b,c] = h_final[b,:] . W_ph[:,c] + b_p[c]  (128x10)
// ---------------------------------------------------------------------------
__global__ __launch_bounds__(256) void rnn_proj(
    const float* __restrict__ h, const float* __restrict__ W_ph,
    const float* __restrict__ b_p, float* __restrict__ out)
{
    int id = blockIdx.x * 256 + threadIdx.x;
    if (id >= BB * CC) return;
    int b = id / CC, cc = id % CC;
    float acc = b_p[cc];
    for (int k = 0; k < HH; ++k)
        acc += h[b * HH + k] * W_ph[k * CC + cc];
    out[id] = acc;
}

extern "C" void kernel_launch(void* const* d_in, const int* in_sizes, int n_in,
                              void* d_out, int out_size, void* d_ws, size_t ws_size,
                              hipStream_t stream)
{
    (void)in_sizes; (void)n_in; (void)out_size; (void)ws_size;
    const float* x    = (const float*)d_in[0];   // [512,128,256]
    const float* W_hx = (const float*)d_in[1];   // [256,512]
    const float* W_hh = (const float*)d_in[2];   // [512,512]
    const float* W_ph = (const float*)d_in[3];   // [512,10]
    const float* b_h  = (const float*)d_in[4];   // [512]
    const float* b_p  = (const float*)d_in[5];   // [10]
    float* out = (float*)d_out;                  // [128,10]

    // Workspace: packed weights (768 KB) + final fp32 h (256 KB)
    char* ws = (char*)d_ws;
    unsigned short* Wfrag = (unsigned short*)ws;
    float* hfinal = (float*)(ws + (size_t)NCHUNK * NTILES * 512 * sizeof(unsigned short));

    // 24*32*512 = 393216 packed weight elements -> 1536 blocks
    rnn_pack_weights<<<1536, 256, 0, stream>>>(W_hx, W_hh, Wfrag);

    // all 512 recurrence steps: 8 independent row-block WGs, zero global sync
    rnn_persist<<<8, 1024, 0, stream>>>(x, Wfrag, b_h, hfinal);

    rnn_proj<<<5, 256, 0, stream>>>(hfinal, W_ph, b_p, out);
}
